// KStore_17008070492704
// MI455X (gfx1250) — matrix operand order repaired and checked by reading, not compiled, not run on hardware
//
#include <hip/hip_runtime.h>

#define NKEYS   500000
#define NQ      256
#define KD      128
#define VD      128
#define TOPK    8
#define NTILES  (NKEYS / 16)                  // 31250 key tiles of 16
#define NB      500                           // chunks / main-kernel blocks
#define TPB_T   ((NTILES + NB - 1) / NB)      // 63 tiles per block
#define NCAND   (NB * TOPK)                   // 4000 candidates per query

typedef __attribute__((ext_vector_type(16))) _Float16 v16h;
typedef __attribute__((ext_vector_type(8)))  float    v8f;

#define NEG_INF (-__builtin_inff())

static __device__ __forceinline__ unsigned pack_h2(float a, float b) {
  union { _Float16 h[2]; unsigned u; } p;
  p.h[0] = (_Float16)a; p.h[1] = (_Float16)b;
  return p.u;
}

// Register-resident top-8 list (sorted descending). Fully unrolled so SROA
// keeps it in VGPRs (no dynamic indexing -> no scratch).
struct Top8 {
  float s[8]; int id[8];
  __device__ __forceinline__ void init() {
#pragma unroll
    for (int j = 0; j < 8; ++j) { s[j] = NEG_INF; id[j] = 0; }
  }
  __device__ __forceinline__ void insert(float v, int i) {
    if (v <= s[7]) return;
    s[7] = v; id[7] = i;
#pragma unroll
    for (int t = 7; t > 0; --t) {
      if (s[t] > s[t - 1]) {
        float tv = s[t]; s[t] = s[t - 1]; s[t - 1] = tv;
        int   ti = id[t]; id[t] = id[t - 1]; id[t - 1] = ti;
      }
    }
  }
};

// ---------------- kernel 1: L2-normalize queries (f32) --------------------
__global__ void __launch_bounds__(32)
qnorm_kernel(const float* __restrict__ q, float* __restrict__ qn) {
  const int row  = blockIdx.x;
  const int lane = threadIdx.x;                       // 0..31, one wave
  const float4 x = *(const float4*)&q[(size_t)row * KD + lane * 4];
  float ss = x.x * x.x + x.y * x.y + x.z * x.z + x.w * x.w;
#pragma unroll
  for (int o = 16; o > 0; o >>= 1) ss += __shfl_xor(ss, o, 32);
  const float rn = 1.0f / fmaxf(sqrtf(ss), 1e-12f);
  float4 y; y.x = x.x * rn; y.y = x.y * rn; y.z = x.z * rn; y.w = x.w * rn;
  *(float4*)&qn[(size_t)row * KD + lane * 4] = y;
}

// ---------------- kernel 2: WMMA sim + per-chunk top-8 --------------------
// 16 waves; wave w owns query tile [16w, 16w+16). Key tiles (16 keys x 128)
// staged in LDS as f16 (double-buffered), key rnorms computed in f32.
__global__ void __launch_bounds__(512)
sim_topk_kernel(const float* __restrict__ keys,
                const float* __restrict__ qn,
                float* __restrict__ cand_s,
                int*   __restrict__ cand_i) {
  // f16 key tile: 16 rows x 128 f16 (=64 uints) padded to 68 uints/row to
  // break the 16-way bank conflict on fragment reads (68 % 64 = 4).
  __shared__ __attribute__((aligned(16))) unsigned tileU[2][16][68];
  __shared__ float norms[2][16];
  // per-wave 16x16 sim transpose scratch, padded row (20 f32, float4-aligned)
  __shared__ __attribute__((aligned(16))) float simS[16][16][20];
  // per-wave merge staging (upper-half lanes publish their lists)
  __shared__ float mS[16][16][8];
  __shared__ int   mI[16][16][8];

  const int tid   = threadIdx.x;
  const int w     = tid >> 5;       // wave id = key row for coop load
  const int lane  = tid & 31;
  const int half  = lane >> 4;      // 0: lanes 0-15, 1: lanes 16-31
  const int l16   = lane & 15;
  const int qbase = w * 16;

  // ---- A fragments: 16x32 f16 per WMMA, per ISA layout:
  // lane holds row M=l16; lower lanes K runs {0..7,16..23}, upper {8..15,24..31}
  const int m   = qbase + l16;
  const int off = half ? 8 : 0;
  v16h afrag[4];
#pragma unroll
  for (int f = 0; f < 4; ++f) {
    const int k1 = 32 * f + off;
    const int k2 = 32 * f + 16 + off;
    const float4 x0 = *(const float4*)&qn[(size_t)m * KD + k1];
    const float4 x1 = *(const float4*)&qn[(size_t)m * KD + k1 + 4];
    const float4 y0 = *(const float4*)&qn[(size_t)m * KD + k2];
    const float4 y1 = *(const float4*)&qn[(size_t)m * KD + k2 + 4];
    union { v16h v; unsigned u[8]; } A;
    A.u[0] = pack_h2(x0.x, x0.y); A.u[1] = pack_h2(x0.z, x0.w);
    A.u[2] = pack_h2(x1.x, x1.y); A.u[3] = pack_h2(x1.z, x1.w);
    A.u[4] = pack_h2(y0.x, y0.y); A.u[5] = pack_h2(y0.z, y0.w);
    A.u[6] = pack_h2(y1.x, y1.y); A.u[7] = pack_h2(y1.z, y1.w);
    afrag[f] = A.v;
  }

  Top8 list; list.init();

  const int t0 = blockIdx.x * TPB_T;
  int ntile = NTILES - t0;
  if (ntile > TPB_T) ntile = TPB_T;
  if (ntile < 0)     ntile = 0;

  // cooperative tile loader: wave w loads key row w (128 f32 = one float4/lane),
  // reduces sum-of-squares across the wave, stores packed f16 row + rnorm.
  auto load_tile = [&](int t, int buf) {
    const int gk = (t0 + t) * 16 + w;
    const float4 kv = *(const float4*)&keys[(size_t)gk * KD + lane * 4];
    float ss = kv.x * kv.x + kv.y * kv.y + kv.z * kv.z + kv.w * kv.w;
#pragma unroll
    for (int o = 16; o > 0; o >>= 1) ss += __shfl_xor(ss, o, 32);
    tileU[buf][w][lane * 2]     = pack_h2(kv.x, kv.y);
    tileU[buf][w][lane * 2 + 1] = pack_h2(kv.z, kv.w);
    if (lane == 0) norms[buf][w] = 1.0f / fmaxf(sqrtf(ss), 1e-12f);
  };

  const int colU = half * 4;   // B fragment: K offset within 16-uint span
  const int cb   = half * 8;   // columns this lane merges for its query

  if (ntile > 0) load_tile(0, 0);
  __syncthreads();

  for (int t = 0; t < ntile; ++t) {
    const int buf = t & 1;
    if (t + 1 < ntile) load_tile(t + 1, buf ^ 1);
    if (t + 2 < ntile)
      __builtin_prefetch(&keys[(size_t)((t0 + t + 2) * 16 + w) * KD + lane * 4], 0, 0);

    // B fragments (32x16 f16): lane = column key l16, lower lanes K {0..7,16..23},
    // upper lanes K {8..15,24..31} of each 32-wide K chunk (matches A layout).
    v16h bfrag[4];
#pragma unroll
    for (int f = 0; f < 4; ++f) {
      union { v16h v; uint4 q2[2]; } B;
      const unsigned* p = &tileU[buf][l16][f * 16 + colU];
      B.q2[0] = *(const uint4*)p;
      B.q2[1] = *(const uint4*)(p + 8);
      bfrag[f] = B.v;
    }

    v8f acc = {};
#pragma unroll
    for (int f = 0; f < 4; ++f)
      acc = __builtin_amdgcn_wmma_f32_16x16x32_f16(
          false, afrag[f], false, bfrag[f], (short)0, acc, false, false);

    // scale column n = l16 by its key rnorm, scatter 16x16 tile to LDS:
    // D layout: VGPR r -> row M = half*8 + r, column N = l16.
    const float rn = norms[buf][l16];
    const int   mb = half * 8;
#pragma unroll
    for (int r = 0; r < 8; ++r)
      simS[w][mb + r][l16] = acc[r] * rn;

    // lane merges its query's (row l16) 8 columns [cb, cb+8) into its top-8.
    // Same-wave LDS ops are in-order; no barrier needed for the transpose.
    const float4 s0 = *(const float4*)&simS[w][l16][cb];
    const float4 s1 = *(const float4*)&simS[w][l16][cb + 4];
    const int nb = (t0 + t) * 16 + cb;
    list.insert(s0.x, nb + 0); list.insert(s0.y, nb + 1);
    list.insert(s0.z, nb + 2); list.insert(s0.w, nb + 3);
    list.insert(s1.x, nb + 4); list.insert(s1.y, nb + 5);
    list.insert(s1.z, nb + 6); list.insert(s1.w, nb + 7);

    __syncthreads();
  }

  // merge the two half-lists per query (same wave), write chunk candidates
  if (half) {
#pragma unroll
    for (int j = 0; j < 8; ++j) { mS[w][l16][j] = list.s[j]; mI[w][l16][j] = list.id[j]; }
  }
  if (!half) {
#pragma unroll
    for (int j = 0; j < 8; ++j) list.insert(mS[w][l16][j], mI[w][l16][j]);
    float* cs = cand_s + ((size_t)(qbase + l16) * NB + blockIdx.x) * TOPK;
    int*   ci = cand_i + ((size_t)(qbase + l16) * NB + blockIdx.x) * TOPK;
#pragma unroll
    for (int j = 0; j < 8; ++j) { cs[j] = list.s[j]; ci[j] = list.id[j]; }
  }
}

// ---------------- kernel 3: global top-8 per query + gather ---------------
__global__ void __launch_bounds__(256)
reduce_gather_kernel(const float* __restrict__ values,
                     const float* __restrict__ cand_s,
                     const int*   __restrict__ cand_i,
                     float* __restrict__ out_slots,
                     float* __restrict__ out_scores) {
  __shared__ float cs[NCAND];
  __shared__ int   ci[NCAND];
  __shared__ float rs[256];
  __shared__ int   rp[256];
  __shared__ int   win[TOPK];

  const int q   = blockIdx.x;
  const int tid = threadIdx.x;

  for (int i = tid; i < NCAND; i += 256) {
    cs[i] = cand_s[(size_t)q * NCAND + i];
    ci[i] = cand_i[(size_t)q * NCAND + i];
  }
  __syncthreads();

  for (int j = 0; j < TOPK; ++j) {
    float best = NEG_INF; int bpos = 0;
    for (int i = tid; i < NCAND; i += 256)
      if (cs[i] > best) { best = cs[i]; bpos = i; }
    rs[tid] = best; rp[tid] = bpos;
    __syncthreads();
    for (int o = 128; o > 0; o >>= 1) {
      if (tid < o && rs[tid + o] > rs[tid]) { rs[tid] = rs[tid + o]; rp[tid] = rp[tid + o]; }
      __syncthreads();
    }
    if (tid == 0) {
      const int p = rp[0];
      out_scores[q * TOPK + j] = rs[0];
      win[j] = ci[p];
      cs[p] = NEG_INF;            // remove winner for next round
    }
    __syncthreads();
  }

  // gather 8 value rows (8 x 128 f32): slot j = tid/32, 4 floats per thread
  const int j = tid >> 5;
  const int c = (tid & 31) * 4;
  const int key = win[j];
  const float4 v = *(const float4*)&values[(size_t)key * VD + c];
  *(float4*)&out_slots[((size_t)q * TOPK + j) * VD + c] = v;
}

// --------------------------------------------------------------------------
extern "C" void kernel_launch(void* const* d_in, const int* in_sizes, int n_in,
                              void* d_out, int out_size, void* d_ws, size_t ws_size,
                              hipStream_t stream) {
  const float* query  = (const float*)d_in[0];
  const float* keys   = (const float*)d_in[1];
  const float* values = (const float*)d_in[2];
  // d_in[3] is k (==8), compile-time constant here.

  float* out        = (float*)d_out;
  float* out_slots  = out;                              // [256][8][128]
  float* out_scores = out + (size_t)NQ * TOPK * VD;     // [256][8]

  // workspace layout: q_norm (128 KB) | cand scores (4 MB) | cand idx (4 MB)
  float* qn     = (float*)d_ws;
  float* cand_s = qn + (size_t)NQ * KD;
  int*   cand_i = (int*)(cand_s + (size_t)NQ * NCAND);

  qnorm_kernel<<<NQ, 32, 0, stream>>>(query, qn);
  sim_topk_kernel<<<NB, 512, 0, stream>>>(keys, qn, cand_s, cand_i);
  reduce_gather_kernel<<<NQ, 256, 0, stream>>>(values, cand_s, cand_i,
                                               out_slots, out_scores);
}
